// StandardMoE_1408749273828
// MI455X (gfx1250) — matrix operand order repaired and checked by reading, not compiled, not run on hardware
//
#include <hip/hip_runtime.h>

// ---------------- problem constants ----------------
#define HID   1024
#define INTER 2048
#define NEXP  8
#define NTOK  8192            // 4 * 2048
#define MT    16              // tokens per tile
#define NC    128             // inter chunk width
#define TPB   512             // 16 waves
#define OUT_ELEMS (NTOK * HID)

// ---------------- WMMA fragment types ----------------
typedef __attribute__((ext_vector_type(16))) __bf16 v16bf;
typedef __attribute__((ext_vector_type(8)))  float  v8f;

union FragB { uint4 q[2]; unsigned int u[8]; v16bf b; };

static __device__ __forceinline__ unsigned short f2bf(float f) {
    unsigned int u = __float_as_uint(f);
    unsigned int r = ((u >> 16) & 1u) + 0x7FFFu;   // round-to-nearest-even
    return (unsigned short)((u + r) >> 16);
}

// ---------------- workspace layout (bytes) ----------------
#define OFF_CNT   0                               // 8 * int
#define OFF_T1    32                              // 8 * float
#define OFF_PS    64                              // 8 * float
#define OFF_LIST  256                             // 8*8192 int  = 262144
#define OFF_GL    (OFF_LIST + NEXP * NTOK * 4)    // 8*8192 float
#define OFF_XBF   (OFF_GL   + NEXP * NTOK * 4)    // NTOK*HID bf16
#define OFF_W1T   (OFF_XBF  + (size_t)NTOK * HID * 2)
#define OFF_W2T   (OFF_W1T  + (size_t)NEXP * HID * INTER * 2)

// ============================================================
// small init: zero counters / stats
// ============================================================
__global__ void init_small_kernel(int* cnt, float* t1, float* ps) {
    int t = threadIdx.x;
    if (t < NEXP) { cnt[t] = 0; t1[t] = 0.f; ps[t] = 0.f; }
}

__global__ void zero_out_kernel(float* out, int n) {
    for (int i = blockIdx.x * blockDim.x + threadIdx.x; i < n; i += gridDim.x * blockDim.x)
        out[i] = 0.f;
}

// ============================================================
// conversions: fp32 -> bf16 (x row-major; W1/W2 transposed)
// ============================================================
__global__ void conv_x_kernel(const float* __restrict__ x, unsigned short* __restrict__ xbf, int n) {
    for (int i = blockIdx.x * blockDim.x + threadIdx.x; i < n; i += gridDim.x * blockDim.x)
        xbf[i] = f2bf(x[i]);
}

// w1 [E][H][I] -> w1t [E][I][H]
__global__ void conv_w1_kernel(const float* __restrict__ w1, unsigned short* __restrict__ w1t) {
    const int n = NEXP * HID * INTER;
    for (int i = blockIdx.x * blockDim.x + threadIdx.x; i < n; i += gridDim.x * blockDim.x) {
        int e   = i / (HID * INTER);
        int rem = i - e * (HID * INTER);
        int h   = rem / INTER;
        int ii  = rem - h * INTER;
        w1t[((size_t)e * INTER + ii) * HID + h] = f2bf(w1[i]);
    }
}

// w2 [E][I][H] -> w2t [E][H][I]
__global__ void conv_w2_kernel(const float* __restrict__ w2, unsigned short* __restrict__ w2t) {
    const int n = NEXP * HID * INTER;
    for (int i = blockIdx.x * blockDim.x + threadIdx.x; i < n; i += gridDim.x * blockDim.x) {
        int e   = i / (HID * INTER);
        int rem = i - e * (HID * INTER);
        int ii  = rem / HID;
        int h   = rem - ii * HID;
        w2t[((size_t)e * HID + h) * INTER + ii] = f2bf(w2[i]);
    }
}

// ============================================================
// router: one wave32 per token
// ============================================================
__global__ __launch_bounds__(256)
void router_kernel(const float* __restrict__ x, const float* __restrict__ wr,
                   int* cnt, float* t1cnt, float* psum,
                   int* lists, float* glists) {
    const int lane = threadIdx.x & 31;
    const int wv   = threadIdx.x >> 5;
    const int token = blockIdx.x * 8 + wv;
    if (token >= NTOK) return;

    float acc[NEXP];
#pragma unroll
    for (int e = 0; e < NEXP; ++e) acc[e] = 0.f;

    const float* xr = x + (size_t)token * HID;
    for (int h = lane; h < HID; h += 32) {
        float xv = xr[h];
        const float* w = wr + h * NEXP;
#pragma unroll
        for (int e = 0; e < NEXP; ++e) acc[e] = fmaf(xv, w[e], acc[e]);
    }
#pragma unroll
    for (int off = 16; off > 0; off >>= 1)
#pragma unroll
        for (int e = 0; e < NEXP; ++e) acc[e] += __shfl_xor(acc[e], off, 32);

    if (lane == 0) {
        float mx = acc[0];
#pragma unroll
        for (int e = 1; e < NEXP; ++e) mx = fmaxf(mx, acc[e]);
        float p[NEXP], s = 0.f;
#pragma unroll
        for (int e = 0; e < NEXP; ++e) { p[e] = __expf(acc[e] - mx); s += p[e]; }
        float inv = 1.f / s;
#pragma unroll
        for (int e = 0; e < NEXP; ++e) p[e] *= inv;

        int i0 = 0;
#pragma unroll
        for (int e = 1; e < NEXP; ++e) if (p[e] > p[i0]) i0 = e;
        int i1 = (i0 == 0) ? 1 : 0;
#pragma unroll
        for (int e = 0; e < NEXP; ++e) if (e != i0 && p[e] > p[i1]) i1 = e;

        float gsum = p[i0] + p[i1];
        float g0 = p[i0] / gsum, g1 = p[i1] / gsum;

        int p0 = atomicAdd(&cnt[i0], 1);
        lists[i0 * NTOK + p0] = token;  glists[i0 * NTOK + p0] = g0;
        int p1 = atomicAdd(&cnt[i1], 1);
        lists[i1 * NTOK + p1] = token;  glists[i1 * NTOK + p1] = g1;

        atomicAdd(&t1cnt[i0], 1.f);
#pragma unroll
        for (int e = 0; e < NEXP; ++e) atomicAdd(&psum[e], p[e]);
    }
}

// ============================================================
// fused expert MLP: silu(X @ W1) @ W2, gated scatter-add
// ============================================================
__global__ __launch_bounds__(TPB)
void moe_expert_kernel(const uint4* __restrict__ xbf4,         // [NTOK][HID/8] uint4
                       const unsigned int* __restrict__ w1t,   // [E][INTER][HID/2] dwords
                       const unsigned int* __restrict__ w2t,   // [E][HID][INTER/2] dwords
                       const int* __restrict__ lists, const float* __restrict__ glists,
                       const int* __restrict__ cnt, float* __restrict__ out) {
    __shared__ unsigned int sXu[MT * HID / 2];   // 32 KB  bf16 X tile
    __shared__ unsigned int sHu[MT * NC  / 2];   // 4 KB   bf16 silu tile
    __shared__ int   sTok[MT];
    __shared__ float sGate[MT];

    const int e  = blockIdx.y;
    const int Ce = cnt[e];
    const int tstart = blockIdx.x * MT;
    if (tstart >= Ce) return;

    const int tid  = threadIdx.x;
    const int lane = tid & 31;
    const int wv   = tid >> 5;
    const int n    = lane & 15;           // col (B/C frag), also row m for A frag
    const int half = lane >> 4;

    if (tid < MT) {
        int idx = tstart + tid;
        if (idx < Ce) { sTok[tid] = lists[e * NTOK + idx]; sGate[tid] = glists[e * NTOK + idx]; }
        else          { sTok[tid] = -1;                     sGate[tid] = 0.f; }
    }
    __syncthreads();

    // gather X rows (bf16), 16B vectors: each row = 128 uint4
    {
        const uint4 z = make_uint4(0u, 0u, 0u, 0u);
        uint4* sX4 = (uint4*)sXu;
        for (int i = tid; i < MT * (HID / 8); i += TPB) {
            int r = i >> 7;               // / 128
            int c = i & 127;
            int tok = sTok[r];
            sX4[i] = (tok >= 0) ? xbf4[(size_t)tok * (HID / 8) + c] : z;
        }
    }
    __syncthreads();

    v8f accT[4] = {};

    // hoisted per-lane fragment base pointers (dword units)
    const unsigned int* w1p = w1t + ((((size_t)(e * INTER + wv * 16 + n)) * HID + (half << 3)) >> 1);
    const unsigned int* w2p = w2t + ((((size_t)(e * HID + wv * 64 + n)) * INTER + (half << 3)) >> 1);

    for (int ic = 0; ic < INTER / NC; ++ic) {
        // ---------------- Phase A: Hc = silu(X @ W1[:, chunk]) ----------------
        if (wv < 8) {
            v8f acc = {};
            const unsigned int* bp = w1p + (size_t)ic * (NC * HID / 2);  // this chunk's col block
            if (ic + 1 < INTER / NC)
                __builtin_prefetch((const char*)(bp + (NC * HID / 2)), 0, 0);
#pragma unroll 4
            for (int kk = 0; kk < HID / 32; ++kk) {
                FragB a, b;
                const int abase = (n * HID + kk * 32 + (half << 3)) >> 1;   // dword idx
                a.q[0] = *(const uint4*)(sXu + abase);
                a.q[1] = *(const uint4*)(sXu + abase + 8);
                const uint4* bq = (const uint4*)(bp + kk * 16);
                b.q[0] = bq[0];
                b.q[1] = bq[2];
                acc = __builtin_amdgcn_wmma_f32_16x16x32_bf16(false, a.b, false, b.b,
                                                              (short)0, acc, false, false);
            }
            unsigned short* sH16 = (unsigned short*)sHu;
#pragma unroll
            for (int v = 0; v < 8; ++v) {                // C layout: row = v + 8*half
                float xv = acc[v];
                float sv = xv / (1.f + __expf(-xv));     // SiLU
                sH16[(v + (half << 3)) * NC + wv * 16 + n] = f2bf(sv);
            }
        }
        __syncthreads();

        // ---------------- Phase B: O += Hc @ W2[chunk, wv*64 .. +64] ----------------
#pragma unroll
        for (int kk = 0; kk < NC / 32; ++kk) {
            FragB a;
            const int abase = (n * NC + kk * 32 + (half << 3)) >> 1;
            a.q[0] = *(const uint4*)(sHu + abase);
            a.q[1] = *(const uint4*)(sHu + abase + 8);
#pragma unroll
            for (int t = 0; t < 4; ++t) {
                // column t*16 further along N; same K offset pattern
                const uint4* bq = (const uint4*)(w2p + (size_t)t * (16 * INTER / 2)
                                                 + (ic * NC + kk * 32) / 2);
                FragB b;
                b.q[0] = bq[0];
                b.q[1] = bq[2];
                accT[t] = __builtin_amdgcn_wmma_f32_16x16x32_bf16(false, a.b, false, b.b,
                                                                  (short)0, accT[t], false, false);
            }
        }
        __syncthreads();
    }

    // ---------------- gated scatter-add ----------------
#pragma unroll
    for (int t = 0; t < 4; ++t)
#pragma unroll
        for (int v = 0; v < 8; ++v) {
            int row = v + (half << 3);
            int tok = sTok[row];
            if (tok >= 0) {
                float g = sGate[row];
                atomicAdd(&out[(size_t)tok * HID + wv * 64 + t * 16 + n], g * accT[t][v]);
            }
        }
}

// ============================================================
// aux loss scalar
// ============================================================
__global__ void aux_kernel(const float* t1cnt, const float* psum, float* out_aux) {
    if (threadIdx.x == 0 && blockIdx.x == 0) {
        float s = 0.f;
        const float invN = 1.f / (float)NTOK;
#pragma unroll
        for (int e = 0; e < NEXP; ++e)
            s += (t1cnt[e] * invN) * (psum[e] * invN);
        *out_aux = s * (float)NEXP;
    }
}

// ============================================================
extern "C" void kernel_launch(void* const* d_in, const int* in_sizes, int n_in,
                              void* d_out, int out_size, void* d_ws, size_t ws_size,
                              hipStream_t stream) {
    (void)in_sizes; (void)n_in; (void)out_size; (void)ws_size;

    const float* x  = (const float*)d_in[0];
    const float* wr = (const float*)d_in[1];
    const float* w1 = (const float*)d_in[2];
    const float* w2 = (const float*)d_in[3];
    float* out = (float*)d_out;

    char* ws = (char*)d_ws;
    int*            cnt    = (int*)(ws + OFF_CNT);
    float*          t1cnt  = (float*)(ws + OFF_T1);
    float*          psum   = (float*)(ws + OFF_PS);
    int*            lists  = (int*)(ws + OFF_LIST);
    float*          glists = (float*)(ws + OFF_GL);
    unsigned short* xbf    = (unsigned short*)(ws + OFF_XBF);
    unsigned short* w1t    = (unsigned short*)(ws + OFF_W1T);
    unsigned short* w2t    = (unsigned short*)(ws + OFF_W2T);

    init_small_kernel<<<1, 32, 0, stream>>>(cnt, t1cnt, psum);
    zero_out_kernel<<<4096, 256, 0, stream>>>(out, OUT_ELEMS);

    conv_x_kernel <<<4096, 256, 0, stream>>>(x, xbf, NTOK * HID);
    conv_w1_kernel<<<8192, 256, 0, stream>>>(w1, w1t);
    conv_w2_kernel<<<8192, 256, 0, stream>>>(w2, w2t);

    router_kernel<<<NTOK / 8, 256, 0, stream>>>(x, wr, cnt, t1cnt, psum, lists, glists);

    dim3 grid(NTOK / MT, NEXP);
    moe_expert_kernel<<<grid, TPB, 0, stream>>>((const uint4*)xbf,
                                                (const unsigned int*)w1t,
                                                (const unsigned int*)w2t,
                                                lists, glists, cnt, out);

    aux_kernel<<<1, 32, 0, stream>>>(t1cnt, psum, out + OUT_ELEMS);
}
